// EnergyBasedModel_47931835023702
// MI455X (gfx1250) — matrix operand order repaired and checked by reading, not compiled, not run on hardware
//
#include <hip/hip_runtime.h>

// ---------------------------------------------------------------------------
// EBM relaxation for MI455X (gfx1250, wave32, WMMA).
// Heavy GEMMs (4096x2048x2048) on v_wmma_f32_16x16x32_bf16, f32 accumulate.
// Wave tile 64x64 (4x4 WMMA tiles): 16 WMMAs per 16 global_load_b128 per
// 32-K step -> ~33 FLOP per L2 byte, compute-bound.  MODE is a template
// parameter so each instantiation has one straight-line epilogue (the
// runtime-mode branch made the allocator spill all accumulators across the
// CFG in rounds 2-4).  States f32; sigmoid activations emitted as bf16 in
// the GEMM epilogues. sig(x)@w0+b0 hoisted out of the 20-step loop.
// ---------------------------------------------------------------------------

typedef __attribute__((ext_vector_type(8)))  float   v8f;
typedef __attribute__((ext_vector_type(8)))  __bf16  v8bf;
typedef __attribute__((ext_vector_type(16))) __bf16  v16bf;

#define STEPSZ 0.005f           // DT / N_STEPS = 0.1 / 20
#define NCOLS  2048
#define BATCH  4096

__device__ __forceinline__ float sigf(float x) { return 1.0f / (1.0f + __expf(-x)); }

// ---------------------------------------------------------------------------
// Pack a KxN f32 matrix into the wave32 WMMA bf16 B-tile layout.
// Tile (kt,nt) covers K rows [32kt,32kt+32), N cols [16nt,16nt+16).
// Within a tile (512 elems): lane holds column (nt*16 + lane&15) and 16
// contiguous K values starting at 32kt + 16*(lane>>4).  Tiles ordered
// (nt-major, kt-minor) so the K sweep streams contiguously.
// transposed==1 means the logical B[k][n] = src[n*K + k].
// ---------------------------------------------------------------------------
__global__ void pack_b_kernel(const float* __restrict__ src, __bf16* __restrict__ dst,
                              int K, int N, int transposed) {
  int idx = blockIdx.x * blockDim.x + threadIdx.x;
  int total = K * N;
  if (idx >= total) return;
  int tile   = idx >> 9;
  int within = idx & 511;
  int lane   = within >> 4;
  int t      = within & 15;
  int Ktiles = K >> 5;
  int nt = tile / Ktiles;
  int kt = tile - nt * Ktiles;
  int n = (nt << 4) + (lane & 15);
  int k = (kt << 5) + ((lane >> 4) << 4) + t;
  float v = transposed ? src[(size_t)n * K + k] : src[(size_t)k * N + n];
  dst[idx] = (__bf16)v;
}

__global__ void sig_to_bf16_kernel(const float* __restrict__ src, __bf16* __restrict__ dst, int n) {
  int i = blockIdx.x * blockDim.x + threadIdx.x;
  if (i < n) dst[i] = (__bf16)sigf(src[i]);
}

__global__ void sig_f32_kernel(const float* __restrict__ src, float* __restrict__ dst, int n) {
  int i = blockIdx.x * blockDim.x + threadIdx.x;
  if (i < n) dst[i] = sigf(src[i]);
}

__global__ void copy_f32_kernel(const float* __restrict__ src, float* __restrict__ dst, int n) {
  int i = blockIdx.x * blockDim.x + threadIdx.x;
  if (i < n) dst[i] = src[i];
}

// ---------------------------------------------------------------------------
// Fused GEMM + state-update kernel.  256 threads = 8 waves (wave32).
// Block tile 128x256 (waves 2M x 4N), wave tile 64x64 = 4x4 WMMA tiles.
// MODE 0: out  = A@B + bias            (C0 precompute; state <- result)
// MODE 1: pre  = A@B + C0;   s1 update; actOut <- sig(s1_new) bf16
// MODE 2: pre  = A@B + bias + sig3@w2T; s2 update; actOut <- sig(s2_new) bf16
// ---------------------------------------------------------------------------
template <int MODE>
__global__ __launch_bounds__(256, 1)
void ebm_gemm_kernel(const __bf16* __restrict__ A, int lda,
                     const __bf16* __restrict__ Bp, int K,
                     const float* __restrict__ C0,
                     const float* __restrict__ bias,
                     float* __restrict__ state,
                     __bf16* __restrict__ actOut,
                     const float* __restrict__ sig3,
                     const float* __restrict__ w2) {
  __shared__ float s3sh[128 * 10];

  const int lane = threadIdx.x & 31;
  const int wv   = threadIdx.x >> 5;
  const int wm   = wv & 1;          // wave M index (0..1)
  const int wn   = wv >> 1;         // wave N index (0..3)
  const int blockM = blockIdx.y;
  const int blockN = blockIdx.x;
  const int waveM  = blockM * 128 + wm * 64;
  const int waveN  = blockN * 256 + wn * 64;
  const int hl   = lane >> 4;       // lane half (0/1)
  const int mrow = lane & 15;

  if constexpr (MODE == 2) {
    for (int t = threadIdx.x; t < 128 * 10; t += 256)
      s3sh[t] = sig3[(size_t)blockM * (128 * 10) + t];
    __syncthreads();
  }

  v8f zero = {0.f, 0.f, 0.f, 0.f, 0.f, 0.f, 0.f, 0.f};
  v8f acc[4][4];
#pragma unroll
  for (int i = 0; i < 4; ++i)
#pragma unroll
    for (int j = 0; j < 4; ++j) acc[i][j] = zero;

  const int Ktiles = K >> 5;
  // Base element offsets (scalar index arithmetic keeps global addrspace).
  // A order used here (same as packed B): lane<16 holds K {0..15}, lane>=16 {16..31}
  size_t aoff0 = (size_t)(waveM + mrow) * lda + hl * 16;
  size_t astride = (size_t)16 * lda;
  // packed B: per-K-step advance is 512 elems; per-lane offset lane*16
  size_t boff0 = ((size_t)(waveN >> 4) * Ktiles) * 512 + lane * 16;
  size_t bstride = (size_t)Ktiles * 512;

  for (int kb = 0; kb < K; kb += 32) {
    v16bf a[4];
#pragma unroll
    for (int i = 0; i < 4; ++i) {
      const __bf16* p = A + aoff0 + (size_t)i * astride + kb;
      v8bf lo = *(const v8bf*)(p);
      v8bf hi = *(const v8bf*)(p + 8);
      a[i] = __builtin_shufflevector(lo, hi, 0, 1, 2, 3, 4, 5, 6, 7,
                                     8, 9, 10, 11, 12, 13, 14, 15);
    }
#pragma unroll
    for (int j = 0; j < 4; ++j) {
      v16bf b = *(const v16bf*)(Bp + boff0 + (size_t)j * bstride + (size_t)kb * 16);
#pragma unroll
      for (int i = 0; i < 4; ++i) {
        acc[i][j] = __builtin_amdgcn_wmma_f32_16x16x32_bf16(false, a[i], false, b,
                                                            (short)0, acc[i][j], false, false);
      }
    }
  }

  // Epilogue. C/D layout: VGPR e, lane<16 -> row base+e col=lane; lane>=16 -> row base+8+e.
  // One accumulator tile at a time; sched_barriers bound scheduler hoisting.
  __builtin_amdgcn_sched_barrier(0);
#pragma unroll
  for (int i = 0; i < 4; ++i) {
#pragma unroll
    for (int j = 0; j < 4; ++j) {
      const int col   = waveN + 16 * j + mrow;
      const int rbase = waveM + 16 * i + hl * 8;
      float w2r[10];
      if constexpr (MODE == 2) {
#pragma unroll
        for (int q = 0; q < 10; ++q) w2r[q] = w2[col * 10 + q];
      }
#pragma unroll
      for (int e = 0; e < 8; ++e) {
        const int row = rbase + e;
        const size_t g = (size_t)row * NCOLS + col;
        float pre = acc[i][j][e];
        if constexpr (MODE == 0) {
          state[g] = pre + bias[col];
        } else if constexpr (MODE == 1) {
          pre += C0[g];
          float s  = state[g];
          float sg = sigf(s);
          float sn = s - STEPSZ * (sg * (1.0f - sg) * pre - s);
          state[g]  = sn;
          actOut[g] = (__bf16)sigf(sn);
        } else {
          pre += bias[col];
          const int lrow = row - blockM * 128;
#pragma unroll
          for (int q = 0; q < 10; ++q) pre += s3sh[lrow * 10 + q] * w2r[q];
          float s  = state[g];
          float sg = sigf(s);
          float sn = s - STEPSZ * (sg * (1.0f - sg) * pre - s);
          state[g]  = sn;
          actOut[g] = (__bf16)sigf(sn);
        }
      }
      __builtin_amdgcn_sched_barrier(0);
    }
  }
}

// ---------------------------------------------------------------------------
// s3 update: pre = sig(s2_new) @ w2 + b2 (K=2048, N=10), one block per row.
// ---------------------------------------------------------------------------
__global__ __launch_bounds__(256)
void s3_update_kernel(const __bf16* __restrict__ act2, const float* __restrict__ w2,
                      const float* __restrict__ b2, float* __restrict__ s3,
                      float* __restrict__ sig3) {
  __shared__ float red[8][10];
  const int row = blockIdx.x;
  const int tid = threadIdx.x;

  float acc[10];
#pragma unroll
  for (int q = 0; q < 10; ++q) acc[q] = 0.0f;

  const __bf16* arow = act2 + (size_t)row * NCOLS;
  for (int k = tid; k < NCOLS; k += 256) {
    float a = (float)arow[k];
#pragma unroll
    for (int q = 0; q < 10; ++q) acc[q] += a * w2[k * 10 + q];
  }
#pragma unroll
  for (int off = 16; off >= 1; off >>= 1) {
#pragma unroll
    for (int q = 0; q < 10; ++q) acc[q] += __shfl_down(acc[q], off, 32);
  }
  const int lane = tid & 31, w = tid >> 5;
  if (lane == 0) {
#pragma unroll
    for (int q = 0; q < 10; ++q) red[w][q] = acc[q];
  }
  __syncthreads();
  if (tid < 10) {
    float pre = b2[tid];
#pragma unroll
    for (int ww = 0; ww < 8; ++ww) pre += red[ww][tid];
    const size_t g = (size_t)row * 10 + tid;
    float s  = s3[g];
    float sg = sigf(s);
    float sn = s - STEPSZ * (sg * (1.0f - sg) * pre - s);
    s3[g]   = sn;
    sig3[g] = sigf(sn);
  }
}

// ---------------------------------------------------------------------------
extern "C" void kernel_launch(void* const* d_in, const int* in_sizes, int n_in,
                              void* d_out, int out_size, void* d_ws, size_t ws_size,
                              hipStream_t stream) {
  const float* x    = (const float*)d_in[0];
  const float* w0   = (const float*)d_in[1];
  const float* w1   = (const float*)d_in[2];
  const float* w2   = (const float*)d_in[3];
  const float* b0   = (const float*)d_in[4];
  const float* b1   = (const float*)d_in[5];
  const float* b2   = (const float*)d_in[6];
  const float* s1in = (const float*)d_in[7];
  const float* s2in = (const float*)d_in[8];
  const float* s3in = (const float*)d_in[9];

  char* ws = (char*)d_ws;
  size_t off = 0;
  auto wsalloc = [&](size_t bytes) {
    char* p = ws + off;
    off += (bytes + 255) & ~(size_t)255;
    return (void*)p;
  };
  float*  S1   = (float*)wsalloc((size_t)BATCH * NCOLS * 4);   // 32 MB
  float*  S2   = (float*)wsalloc((size_t)BATCH * NCOLS * 4);   // 32 MB
  float*  C0   = (float*)wsalloc((size_t)BATCH * NCOLS * 4);   // 32 MB
  __bf16* ACT1 = (__bf16*)wsalloc((size_t)BATCH * NCOLS * 2);  // 16 MB
  __bf16* ACT2 = (__bf16*)wsalloc((size_t)BATCH * NCOLS * 2);  // 16 MB
  __bf16* W1P  = (__bf16*)wsalloc((size_t)2048 * 2048 * 2);    // 8 MB
  __bf16* W1TP = (__bf16*)wsalloc((size_t)2048 * 2048 * 2);    // 8 MB
  float*  S3   = (float*)wsalloc((size_t)BATCH * 10 * 4);
  float*  SG3  = (float*)wsalloc((size_t)BATCH * 10 * 4);
  // xb / packed w0 are only needed before step 0 -> alias into ACT1 (16 MB)
  __bf16* XB  = ACT1;                           // 4096x1024 bf16 = 8 MB
  __bf16* W0P = ACT1 + (size_t)BATCH * 1024;    // 1024x2048 bf16 = 4 MB

  const int TB = 256;
  // one-time packs / conversions / state copies
  pack_b_kernel<<<(1024 * 2048 + TB - 1) / TB, TB, 0, stream>>>(w0, W0P, 1024, 2048, 0);
  pack_b_kernel<<<(2048 * 2048 + TB - 1) / TB, TB, 0, stream>>>(w1, W1P, 2048, 2048, 0);
  pack_b_kernel<<<(2048 * 2048 + TB - 1) / TB, TB, 0, stream>>>(w1, W1TP, 2048, 2048, 1);
  sig_to_bf16_kernel<<<(BATCH * 1024 + TB - 1) / TB, TB, 0, stream>>>(x, XB, BATCH * 1024);
  copy_f32_kernel<<<(BATCH * NCOLS + TB - 1) / TB, TB, 0, stream>>>(s1in, S1, BATCH * NCOLS);
  copy_f32_kernel<<<(BATCH * NCOLS + TB - 1) / TB, TB, 0, stream>>>(s2in, S2, BATCH * NCOLS);
  copy_f32_kernel<<<(BATCH * 10 + TB - 1) / TB, TB, 0, stream>>>(s3in, S3, BATCH * 10);
  sig_to_bf16_kernel<<<(BATCH * NCOLS + TB - 1) / TB, TB, 0, stream>>>(s2in, ACT2, BATCH * NCOLS);
  sig_f32_kernel<<<(BATCH * 10 + TB - 1) / TB, TB, 0, stream>>>(s3in, SG3, BATCH * 10);

  dim3 grid(NCOLS / 256, BATCH / 128);  // (8, 32)
  dim3 blk(256);
  // loop-invariant: C0 = sig(x) @ w0 + b0
  ebm_gemm_kernel<0><<<grid, blk, 0, stream>>>(XB, 1024, W0P, 1024,
                                               nullptr, b0, C0, nullptr, nullptr, nullptr);
  for (int step = 0; step < 20; ++step) {
    // s1 <- update(s1, C0 + sig(s2_old) @ w1^T); ACT1 <- sig(s1_new)
    ebm_gemm_kernel<1><<<grid, blk, 0, stream>>>(ACT2, 2048, W1TP, 2048,
                                                 C0, nullptr, S1, ACT1, nullptr, nullptr);
    // s2 <- update(s2, b1 + sig(s1_new) @ w1 + sig(s3_old) @ w2^T); ACT2 <- sig(s2_new)
    ebm_gemm_kernel<2><<<grid, blk, 0, stream>>>(ACT1, 2048, W1P, 2048,
                                                 nullptr, b1, S2, ACT2, SG3, w2);
    // s3 <- update(s3, b2 + sig(s2_new) @ w2); SG3 <- sig(s3_new)
    s3_update_kernel<<<BATCH, blk, 0, stream>>>(ACT2, w2, b2, S3, SG3);
  }
  copy_f32_kernel<<<(BATCH * 10 + TB - 1) / TB, TB, 0, stream>>>(S3, (float*)d_out, BATCH * 10);
}
